// _NonLocalNd_52286931862026
// MI455X (gfx1250) — matrix operand order repaired and checked by reading, not compiled, not run on hardware
//
#include <hip/hip_runtime.h>
#include <hip/hip_bf16.h>
#include <math.h>

#define NB   8
#define C_   512
#define CI_  256
#define HW_  4096

typedef __attribute__((ext_vector_type(16))) __bf16 v16bf;
typedef __attribute__((ext_vector_type(8)))  __bf16 v8bf;
typedef __attribute__((ext_vector_type(8)))  float  v8f;

#define WMMA_BF16(a, b, c) \
  __builtin_amdgcn_wmma_f32_16x16x32_bf16(false, (a), false, (b), (short)0, (c), false, false)

// ---------------------------------------------------------------------------
// TDM support (Tensor Data Mover). 5-arg builtin on ROCm 7.2 clang-22,
// 6-arg on amdgpu-toolchain clang-23 (which ships the TDM header).
#if __has_builtin(__builtin_amdgcn_tensor_load_to_lds)
#define ATTN_TDM 1
typedef unsigned int u32x4 __attribute__((ext_vector_type(4)));
typedef int          i32x4 __attribute__((ext_vector_type(4)));
typedef int          i32x8 __attribute__((ext_vector_type(8)));
#if __has_include(<hip/amd_detail/amd_gfx1250_TDM.h>)
#define TDM_CALL(g0, g1, za, zb) \
  __builtin_amdgcn_tensor_load_to_lds((g0), (g1), (za), (zb), (i32x8){}, 0)
#else
#define TDM_CALL(g0, g1, za, zb) \
  __builtin_amdgcn_tensor_load_to_lds((g0), (g1), (za), (zb), 0)
#endif

// 2D TDM load: tile_d1 rows x tile_d0 elems (bf16), row stride stride0 elems,
// with LDS pad-on-write (pad_amount DWORDs every pad_interval DWORDs) to kill
// bank conflicts on the B-fragment reads.
__device__ __forceinline__ void tdm_load_2d(unsigned lds_byte_off, unsigned long long gaddr,
                                            unsigned tile_d0, unsigned tile_d1,
                                            unsigned long long tdim0, unsigned long long tdim1,
                                            unsigned long long stride0,
                                            unsigned pad_interval_code, unsigned pad_amount_code) {
  u32x4 g0;
  g0[0] = 1u;                                                   // count=1, user descriptor
  g0[1] = lds_byte_off;                                         // lds_addr
  g0[2] = (unsigned)gaddr;                                      // global_addr[31:0]
  g0[3] = (unsigned)((gaddr >> 32) & 0x01ffffffu) | (2u << 30); // addr[56:32] | type=2
  i32x8 g1;
  g1[0] = (int)((1u << 16)                                      // data_size=1 -> 2 bytes
              | (1u << 20)                                      // pad_enable
              | (pad_interval_code << 22) | (pad_amount_code << 25));
  g1[1] = (int)((tdim0 & 0xffffu) << 16);                       // tensor_dim0[15:0]
  g1[2] = (int)(((tdim0 >> 16) & 0xffffu) | ((tdim1 & 0xffffu) << 16));
  g1[3] = (int)(((tdim1 >> 16) & 0xffffu) | ((tile_d0 & 0xffffu) << 16));
  g1[4] = (int)(tile_d1 & 0xffffu);                             // tile_dim1 (tile_dim2=0)
  g1[5] = (int)(unsigned)(stride0 & 0xffffffffu);               // tensor_dim0_stride[31:0]
  g1[6] = (int)((stride0 >> 32) & 0xffffu);                     // stride[47:32] (dim1_stride=0)
  g1[7] = 0;
  i32x4 z = {};
  TDM_CALL(g0, g1, z, z);
}
__device__ __forceinline__ unsigned lds_off(const void* p) {
  return (unsigned)(size_t)p;   // generic LDS pointer: addr[31:0] = LDS byte offset
}
#else
#define ATTN_TDM 0
#endif

// Padded LDS row strides (elements) chosen so 16 B-frag lanes hit disjoint banks.
#define PHI_STR 264   // 256 + 8  (pad 4 DW every 128 DW)
#define G_STR   40    // 32 + 8   (pad 4 DW every 16 DW)
#define PB_STR  40

// ---------------------------------------------------------------------------
// Fragment loaders (wave32, 16x16x32 bf16 WMMA).
// A 16x32 (MxK): lane l -> m=l&15, half=l>>4; elem j: K=half*16+j.
__device__ __forceinline__ v16bf load_frag_a(const __bf16* base, int stride) {
  int l = threadIdx.x & 31;
  int m = l & 15, h = l >> 4;
  const __bf16* p = base + (size_t)m * stride + h * 16;
  v8bf lo = *(const v8bf*)(p);
  v8bf hi = *(const v8bf*)(p + 8);
  return __builtin_shufflevector(lo, hi, 0,1,2,3,4,5,6,7,8,9,10,11,12,13,14,15);
}
// B 32x16 (KxN): lane l -> n=l&15, half=l>>4; elem j: K=half*16+j. Source [n][k], K contiguous.
__device__ __forceinline__ v16bf load_frag_b(const __bf16* base, int stride) {
  int l = threadIdx.x & 31;
  int n = l & 15, h = l >> 4;
  const __bf16* p = base + (size_t)n * stride + h * 16;
  v8bf lo = *(const v8bf*)(p);
  v8bf hi = *(const v8bf*)(p + 8);
  return __builtin_shufflevector(lo, hi, 0,1,2,3,4,5,6,7,8,9,10,11,12,13,14,15);
}
// C/D 16x16 f32: element r of lane l -> row M = r + 8*(l>>4), col N = l&15.

// ---------------------------------------------------------------------------
__global__ void k_cvt_bf16(const float* __restrict__ src, __bf16* __restrict__ dst, int n) {
  int i = blockIdx.x * blockDim.x + threadIdx.x;
  if (i < n) dst[i] = (__bf16)src[i];
}

__global__ void k_transpose_x(const float* __restrict__ x, __bf16* __restrict__ xT) {
  __shared__ float tile[32][33];
  int n  = blockIdx.z;
  int q0 = blockIdx.x * 32;
  int c0 = blockIdx.y * 32;
  int tx = threadIdx.x, ty = threadIdx.y;
  const float* xs = x + (size_t)n * C_ * HW_;
  #pragma unroll
  for (int i = ty; i < 32; i += 8)
    tile[i][tx] = xs[(size_t)(c0 + i) * HW_ + q0 + tx];
  __syncthreads();
  __bf16* xd = xT + (size_t)n * HW_ * C_;
  #pragma unroll
  for (int i = ty; i < 32; i += 8)
    xd[(size_t)(q0 + i) * C_ + c0 + tx] = (__bf16)tile[tx][i];
}

// ---------------------------------------------------------------------------
// Projection GEMM: proj[o][q] = sum_c w[o][c]*x[n][c][q] + bias[o]; one wave per 16x16 tile.
template <int TRANSPOSE_OUT>
__global__ void k_proj(const __bf16* __restrict__ wbf, const __bf16* __restrict__ xT,
                       const float* __restrict__ bias, __bf16* __restrict__ out) {
  int wave = blockIdx.x * 4 + (threadIdx.x >> 5);
  int qt  = wave & (HW_ / 16 - 1);
  int tmp = wave / (HW_ / 16);
  int ot  = tmp & (CI_ / 16 - 1);
  int n   = tmp / (CI_ / 16);
  const __bf16* xn = xT + (size_t)n * HW_ * C_ + (size_t)(qt * 16) * C_;
  v8f acc = {};
  #pragma unroll
  for (int k = 0; k < C_; k += 32) {
    v16bf a = load_frag_a(wbf + (size_t)(ot * 16) * C_ + k, C_);
    v16bf b = load_frag_b(xn + k, C_);
    acc = WMMA_BF16(a, b, acc);
  }
  int l = threadIdx.x & 31;
  int nn = l & 15, h = l >> 4;
  #pragma unroll
  for (int r = 0; r < 8; ++r) {
    int o = ot * 16 + r + 8 * h;
    int q = qt * 16 + nn;
    float v = acc[r] + bias[o];
    if (TRANSPOSE_OUT)
      out[(size_t)n * HW_ * CI_ + (size_t)q * CI_ + o] = (__bf16)v;
    else
      out[(size_t)n * CI_ * HW_ + (size_t)o * HW_ + q] = (__bf16)v;
  }
}

// ---------------------------------------------------------------------------
// Fused flash attention. 4 waves/block share one batch; wave0 drives TDM
// double-buffering of the phi (32x256) and g (256x32) key tiles into LDS.
__global__ void __launch_bounds__(128)
k_attention(const __bf16* __restrict__ thetaT, const __bf16* __restrict__ phiT,
            const __bf16* __restrict__ g, __bf16* __restrict__ y) {
#if ATTN_TDM
  extern __shared__ __bf16 smem[];
  __bf16* sphi = smem;                          // 2 x [32][PHI_STR]
  __bf16* sg   = smem + 2 * 32 * PHI_STR;       // 2 x [256][G_STR]
  __bf16* pbase = sg + 2 * CI_ * G_STR;         // 4 x [16][PB_STR]
#else
  __shared__ __align__(16) __bf16 pstat[4][16 * PB_STR];
#endif
  int wv   = threadIdx.x >> 5;
  int qt = (blockIdx.x & 63) * 4 + wv;
  int n  = blockIdx.x >> 6;
  int l  = threadIdx.x & 31;
  int nn = l & 15, h = l >> 4;

  const __bf16* thn = thetaT + (size_t)n * HW_ * CI_ + (size_t)(qt * 16) * CI_;
  const __bf16* phn = phiT   + (size_t)n * HW_ * CI_;
  const __bf16* gn  = g      + (size_t)n * CI_ * HW_;
#if ATTN_TDM
  __bf16* pb = pbase + wv * (16 * PB_STR);
  unsigned long long phg = (unsigned long long)(size_t)phn;
  unsigned long long gg  = (unsigned long long)(size_t)gn;
#else
  __bf16* pb = pstat[wv];
#endif

  v16bf aTh[8];                       // 16q x 256ci query tile in VGPRs
  #pragma unroll
  for (int t = 0; t < 8; ++t) aTh[t] = load_frag_a(thn + t * 32, CI_);

  v8f zero = {};
  v8f acc[16];
  #pragma unroll
  for (int t = 0; t < 16; ++t) acc[t] = zero;
  float mrow[8], lrow[8];
  #pragma unroll
  for (int r = 0; r < 8; ++r) { mrow[r] = -3.0e38f; lrow[r] = 0.f; }

#if ATTN_TDM
  if (wv == 0) {   // prologue: fetch tile 0 into buffer 0
    tdm_load_2d(lds_off(sphi), phg, CI_, 32, CI_, HW_, CI_, 6u, 3u); // pad 4DW/128DW
    tdm_load_2d(lds_off(sg),   gg,  32, CI_, HW_, CI_, HW_, 3u, 3u); // pad 4DW/16DW
  }
#endif

  for (int k0 = 0; k0 < HW_; k0 += 32) {
#if ATTN_TDM
    int cur = (k0 >> 5) & 1;
    if (wv == 0) __builtin_amdgcn_s_wait_tensorcnt(0);  // tile `cur` landed
    __syncthreads();                                    // visible to all; prev compute done
    if (wv == 0 && (k0 + 32) < HW_) {
      int nxt = cur ^ 1;
      tdm_load_2d(lds_off(sphi + nxt * 32 * PHI_STR), phg + (size_t)(k0 + 32) * CI_ * 2,
                  CI_, 32, CI_, HW_, CI_, 6u, 3u);
      tdm_load_2d(lds_off(sg + nxt * CI_ * G_STR), gg + (size_t)(k0 + 32) * 2,
                  32, CI_, HW_, CI_, HW_, 3u, 3u);
    }
    const __bf16* phL = sphi + cur * 32 * PHI_STR;   // [key][ci], stride PHI_STR
    const __bf16* gL  = sg   + cur * CI_ * G_STR;    // [ci][key], stride G_STR
    const int phStr = PHI_STR, gStr = G_STR;
#else
    const __bf16* phL = phn + (size_t)k0 * CI_;
    const __bf16* gL  = gn + k0;
    const int phStr = CI_, gStr = HW_;
#endif
    // S = theta . phi^T : 16q x 32keys, K=256
    v8f s0 = zero, s1 = zero;
    #pragma unroll
    for (int t = 0; t < 8; ++t) {
      v16bf b0 = load_frag_b(phL + t * 32, phStr);
      v16bf b1 = load_frag_b(phL + 16 * phStr + t * 32, phStr);
      s0 = WMMA_BF16(aTh[t], b0, s0);
      s1 = WMMA_BF16(aTh[t], b1, s1);
    }
    // online softmax (row M = r+8*h; cols on 16 lanes/half; xor 1/2/4/8 stays in half)
    float p0[8], p1[8], sc[8];
    #pragma unroll
    for (int r = 0; r < 8; ++r) {
      float a = s0[r] * 0.0625f;      // 1/sqrt(CI)
      float b = s1[r] * 0.0625f;
      float mx = fmaxf(a, b);
      mx = fmaxf(mx, __shfl_xor(mx, 1, 32));
      mx = fmaxf(mx, __shfl_xor(mx, 2, 32));
      mx = fmaxf(mx, __shfl_xor(mx, 4, 32));
      mx = fmaxf(mx, __shfl_xor(mx, 8, 32));
      float newm  = fmaxf(mrow[r], mx);
      float scale = __expf(mrow[r] - newm);
      p0[r] = __expf(a - newm);
      p1[r] = __expf(b - newm);
      float rs = p0[r] + p1[r];
      rs += __shfl_xor(rs, 1, 32);
      rs += __shfl_xor(rs, 2, 32);
      rs += __shfl_xor(rs, 4, 32);
      rs += __shfl_xor(rs, 8, 32);
      lrow[r] = lrow[r] * scale + rs;
      mrow[r] = newm;
      sc[r]   = scale;
    }
    #pragma unroll
    for (int t = 0; t < 16; ++t) {
      #pragma unroll
      for (int r = 0; r < 8; ++r) acc[t][r] *= sc[r];
    }
    // P (C/D frag) -> A frag via per-wave LDS staging (same-wave DS ops are in-order)
    #pragma unroll
    for (int r = 0; r < 8; ++r) {
      pb[(r + 8 * h) * PB_STR + nn]      = (__bf16)p0[r];
      pb[(r + 8 * h) * PB_STR + 16 + nn] = (__bf16)p1[r];
    }
    v16bf pa = load_frag_a(pb, PB_STR);
    #pragma unroll
    for (int t = 0; t < 16; ++t) {
      v16bf bg = load_frag_b(gL + (size_t)(t * 16) * gStr, gStr);
      acc[t] = WMMA_BF16(pa, bg, acc[t]);
    }
  }
  float inv[8];
  #pragma unroll
  for (int r = 0; r < 8; ++r) inv[r] = 1.0f / lrow[r];
  __bf16* yn = y + (size_t)n * HW_ * CI_ + (size_t)(qt * 16) * CI_;
  #pragma unroll
  for (int t = 0; t < 16; ++t) {
    #pragma unroll
    for (int r = 0; r < 8; ++r)
      yn[(size_t)(r + 8 * h) * CI_ + t * 16 + nn] = (__bf16)(acc[t][r] * inv[r]);
  }
}

// ---------------------------------------------------------------------------
__global__ void k_outproj(const __bf16* __restrict__ wo, const __bf16* __restrict__ y,
                          const float* __restrict__ ob, const float* __restrict__ x,
                          float* __restrict__ out) {
  int wave = blockIdx.x * 4 + (threadIdx.x >> 5);
  int qt  = wave & (HW_ / 16 - 1);
  int tmp = wave / (HW_ / 16);
  int ct  = tmp & (C_ / 16 - 1);
  int n   = tmp / (C_ / 16);
  const __bf16* yn = y + (size_t)n * HW_ * CI_ + (size_t)(qt * 16) * CI_;
  v8f acc = {};
  #pragma unroll
  for (int k = 0; k < CI_; k += 32) {
    v16bf a = load_frag_a(wo + (size_t)(ct * 16) * CI_ + k, CI_);
    v16bf b = load_frag_b(yn + k, CI_);
    acc = WMMA_BF16(a, b, acc);
  }
  int l = threadIdx.x & 31;
  int nn = l & 15, h = l >> 4;
  #pragma unroll
  for (int r = 0; r < 8; ++r) {
    int c = ct * 16 + r + 8 * h;
    int q = qt * 16 + nn;
    size_t idx = ((size_t)n * C_ + c) * HW_ + q;
    out[idx] = x[idx] + acc[r] + ob[c];
  }
}

// ---------------------------------------------------------------------------
extern "C" void kernel_launch(void* const* d_in, const int* in_sizes, int n_in,
                              void* d_out, int out_size, void* d_ws, size_t ws_size,
                              hipStream_t stream) {
  (void)in_sizes; (void)n_in; (void)out_size; (void)ws_size;
  const float* x    = (const float*)d_in[0];
  const float* g_w  = (const float*)d_in[1];
  const float* g_b  = (const float*)d_in[2];
  const float* th_w = (const float*)d_in[3];
  const float* th_b = (const float*)d_in[4];
  const float* ph_w = (const float*)d_in[5];
  const float* ph_b = (const float*)d_in[6];
  const float* o_w  = (const float*)d_in[7];
  const float* o_b  = (const float*)d_in[8];
  float* out = (float*)d_out;

  char* ws = (char*)d_ws;
  size_t off = 0;
  auto wsalloc = [&](size_t bytes) {
    char* p = ws + off;
    off += (bytes + 255) & ~(size_t)255;
    return p;
  };
  __bf16* xT   = (__bf16*)wsalloc((size_t)NB * HW_ * C_ * 2);   // [n][q][c]
  __bf16* thT  = (__bf16*)wsalloc((size_t)NB * HW_ * CI_ * 2);  // [n][q][ci]
  __bf16* phT  = (__bf16*)wsalloc((size_t)NB * HW_ * CI_ * 2);  // [n][q][ci]
  __bf16* gx   = (__bf16*)wsalloc((size_t)NB * CI_ * HW_ * 2);  // [n][ci][q]
  __bf16* ybf  = (__bf16*)wsalloc((size_t)NB * HW_ * CI_ * 2);  // [n][q][ci]
  __bf16* gwb  = (__bf16*)wsalloc((size_t)CI_ * C_ * 2);
  __bf16* thwb = (__bf16*)wsalloc((size_t)CI_ * C_ * 2);
  __bf16* phwb = (__bf16*)wsalloc((size_t)CI_ * C_ * 2);
  __bf16* owb  = (__bf16*)wsalloc((size_t)C_ * CI_ * 2);

  k_cvt_bf16<<<(CI_ * C_ + 255) / 256, 256, 0, stream>>>(g_w,  gwb,  CI_ * C_);
  k_cvt_bf16<<<(CI_ * C_ + 255) / 256, 256, 0, stream>>>(th_w, thwb, CI_ * C_);
  k_cvt_bf16<<<(CI_ * C_ + 255) / 256, 256, 0, stream>>>(ph_w, phwb, CI_ * C_);
  k_cvt_bf16<<<(C_ * CI_ + 255) / 256, 256, 0, stream>>>(o_w,  owb,  C_ * CI_);
  dim3 tb(32, 8, 1), tg(HW_ / 32, C_ / 32, NB);
  k_transpose_x<<<tg, tb, 0, stream>>>(x, xT);

  int projWaves = NB * (CI_ / 16) * (HW_ / 16);
  k_proj<1><<<projWaves / 4, 128, 0, stream>>>(thwb, xT, th_b, thT);
  k_proj<1><<<projWaves / 4, 128, 0, stream>>>(phwb, xT, ph_b, phT);
  k_proj<0><<<projWaves / 4, 128, 0, stream>>>(gwb,  xT, g_b,  gx);

  int attBlocks = NB * (HW_ / 16) / 4;   // 4 q-tiles per block, same batch
#if ATTN_TDM
  size_t attSmem = (size_t)(2 * 32 * PHI_STR + 2 * CI_ * G_STR + 4 * 16 * PB_STR) * 2;
#else
  size_t attSmem = 0;
#endif
  k_attention<<<attBlocks, 128, attSmem, stream>>>(thT, phT, gx, ybf);

  int outWaves = NB * (C_ / 16) * (HW_ / 16);
  k_outproj<<<outWaves / 4, 128, 0, stream>>>(owb, ybf, o_b, x, out);
}
